// SpatialEncoderLSTM_50148038148182
// MI455X (gfx1250) — compile-verified, matrix operand order checked by reading, not run on hardware
//
#include <hip/hip_runtime.h>
#include <hip/hip_bf16.h>
#include <stdint.h>

// ---------------------------------------------------------------------------
// SpatialEncoderLSTM for MI455X (gfx1250, wave32, WMMA bf16->f32)
//
// Pipeline:
//   prep:        cast W_ih/W_hh -> bf16, bias = b_ih+b_hh, h/c state init
//   embed_gemm:  [51200 x 77] @ W_emb^T + b_emb -> bf16 embedded [A*B, H]
//   per step a (100x, sequential):
//     lstm_gate_gemm: gates[512,4096] = [x_a | h] (K=2048, bf16) @ [W_ih;W_hh]^T
//                     v_wmma_f32_16x16x32_bf16, 128x128 tiles,
//                     async global->LDS double-buffered K pipeline (ASYNCcnt)
//     lstm_pointwise: i/f/g/o -> c,h update (f32), writes output slice
// ---------------------------------------------------------------------------

typedef __bf16 bf16;
typedef __attribute__((ext_vector_type(16))) __bf16 v16bf;
typedef __attribute__((ext_vector_type(8)))  __bf16 v8bf;
typedef __attribute__((ext_vector_type(8)))  float  v8f;

#define DEV __device__ __forceinline__

constexpr int Bn = 512;     // batch
constexpr int An = 100;     // agents (time steps of scan)
constexpr int Hn = 1024;    // hidden
constexpr int INn = 77;     // S*T
constexpr int G4 = 4 * Hn;  // 4096 (i,f,g,o)

// -------------------- WMMA helpers --------------------

DEV v8f wmma_bf16(v16bf a, v16bf b, v8f c) {
  // D = A(16x32 bf16) x B(32x16 bf16) + C(16x16 f32)
  return __builtin_amdgcn_wmma_f32_16x16x32_bf16(
      /*neg_a=*/false, a, /*neg_b=*/false, b,
      /*c_mod=*/(short)0, c, /*reuse_a=*/false, /*reuse_b=*/false);
}

// Load a 16x32 bf16 fragment from LDS (row-major, padded stride).
// ISA 7.12.2 16-bit operand layout:
//   lanes 0-15 : row = lane,    K = {0..7} U {16..23}
//   lanes 16-31: row = lane-16, K = {8..15} U {24..31}
// Two 16B chunks per lane -> ds_load_b128 pairs.
DEV v16bf frag_ld(const bf16* p0 /* base + row0*stride + col0 */, int stride,
                  int lane) {
  const int idx = lane & 15;
  const int koff = (lane >> 4) ? 8 : 0;
  const bf16* p = p0 + idx * stride + koff;
  v8bf lo = *(const v8bf*)(p);
  v8bf hi = *(const v8bf*)(p + 16);
  return __builtin_shufflevector(lo, hi, 0, 1, 2, 3, 4, 5, 6, 7,
                                 8, 9, 10, 11, 12, 13, 14, 15);
}

// Async global->LDS 16B copy (GLOBAL_LOAD_ASYNC_TO_LDS_B128, GV mode):
// vdst = per-lane LDS byte offset, vaddr = per-lane 64-bit global address.
// Tracked by ASYNCcnt; no VGPR data round trip. Inline asm (portable across
// ROCm 7.2 / clang-23 — avoids the TDM builtin arity mismatch).
DEV void async_cp16(uint32_t lds_off, const void* gptr) {
  asm volatile("global_load_async_to_lds_b128 %0, %1, off"
               :
               : "v"(lds_off), "v"((uint64_t)(uintptr_t)gptr)
               : "memory");
}
DEV void wait_async0() {
  asm volatile("s_wait_asynccnt 0" ::: "memory");
}

// -------------------- prep: casts + state init --------------------

__global__ void prep_kernel(const float* __restrict__ Wih,
                            const float* __restrict__ Whh,
                            const float* __restrict__ bih,
                            const float* __restrict__ bhh,
                            const float* __restrict__ h0,
                            const float* __restrict__ c0,
                            bf16* __restrict__ Wihb, bf16* __restrict__ Whhb,
                            float* __restrict__ bias, bf16* __restrict__ hbf,
                            float* __restrict__ c) {
  size_t idx = (size_t)blockIdx.x * blockDim.x + threadIdx.x;
  if (idx < (size_t)G4 * Hn) {
    Wihb[idx] = (bf16)Wih[idx];
    Whhb[idx] = (bf16)Whh[idx];
  }
  if (idx < (size_t)G4) bias[idx] = bih[idx] + bhh[idx];
  if (idx < (size_t)Bn * Hn) {
    hbf[idx] = (bf16)h0[idx];
    c[idx] = c0[idx];
  }
}

// -------------------- embedding GEMM --------------------
// emb[m, n] = sum_i x[m, i] * W_emb[n, i] + b_emb[n],  m = a*512 + b
// K = 77 padded to 96 (3 WMMA k-steps). Single LDS stage (K is tiny).

constexpr int ESTR = 104;  // padded LDS row stride (bf16 elems), 208B = 13*16B

__global__ __launch_bounds__(256) void embed_gemm(
    const float* __restrict__ xin,   // [512, 7700] = [B, A*77]
    const float* __restrict__ Wemb,  // [1024, 77]
    const float* __restrict__ bemb,  // [1024]
    bf16* __restrict__ emb) {        // [A*B, 1024]
  __shared__ bf16 sX[128 * ESTR];
  __shared__ bf16 sW[128 * ESTR];

  const int tid = threadIdx.x;
  const int lane = tid & 31;
  const int wave = tid >> 5;
  const int nt = blockIdx.x & 7;    // 8 N tiles (1024/128)
  const int mt = blockIdx.x >> 3;   // 400 M tiles (51200/128)
  const int m0 = mt * 128, n0 = nt * 128;

  // Stage operands into LDS, converting f32 -> bf16 with zero pad K=77..95.
  {
    const int row = tid >> 1;   // 0..127
    const int hs = tid & 1;     // low/high half of the 96 columns
    const int a = m0 >> 9;      // 512 rows per agent; tiles never straddle a
    const int brow = (m0 & 511) + row;
    const float* src = xin + (size_t)brow * (An * INn) + (size_t)a * INn;
    const float* wsrc = Wemb + (size_t)(n0 + row) * INn;
#pragma unroll
    for (int i = 0; i < 48; ++i) {
      const int col = hs * 48 + i;
      const float xv = (col < INn) ? src[col] : 0.0f;
      const float wv = (col < INn) ? wsrc[col] : 0.0f;
      sX[row * ESTR + col] = (bf16)xv;
      sW[row * ESTR + col] = (bf16)wv;
    }
  }
  __syncthreads();

  const int wm = (wave & 3) * 32;   // wave M offset in tile
  const int wn = (wave >> 2) * 64;  // wave N offset in tile
  const v8f zero8 = {0.f, 0.f, 0.f, 0.f, 0.f, 0.f, 0.f, 0.f};
  v8f acc[2][4];
#pragma unroll
  for (int i = 0; i < 2; ++i)
#pragma unroll
    for (int j = 0; j < 4; ++j) acc[i][j] = zero8;

#pragma unroll
  for (int ks = 0; ks < 3; ++ks) {
    const int c0 = ks * 32;
    v16bf af[2], bv[4];
#pragma unroll
    for (int i = 0; i < 2; ++i)
      af[i] = frag_ld(&sX[(wm + i * 16) * ESTR + c0], ESTR, lane);
#pragma unroll
    for (int j = 0; j < 4; ++j)
      bv[j] = frag_ld(&sW[(wn + j * 16) * ESTR + c0], ESTR, lane);
#pragma unroll
    for (int i = 0; i < 2; ++i)
#pragma unroll
      for (int j = 0; j < 4; ++j) acc[i][j] = wmma_bf16(af[i], bv[j], acc[i][j]);
  }

  // Epilogue: +b_emb, cast to bf16. C layout: VGPR r -> M=r+8*half, N=lane&15.
  const int mhalf = (lane >> 4) ? 8 : 0;
#pragma unroll
  for (int i = 0; i < 2; ++i) {
    const int rbase = m0 + wm + i * 16 + mhalf;
#pragma unroll
    for (int j = 0; j < 4; ++j) {
      const int col = n0 + wn + j * 16 + (lane & 15);
      const float bb = bemb[col];
#pragma unroll
      for (int r = 0; r < 8; ++r)
        emb[(size_t)(rbase + r) * Hn + col] = (bf16)(acc[i][j][r] + bb);
    }
  }
}

// -------------------- per-step gate GEMM --------------------
// gates[512, 4096] = [x_a | h] @ [W_ih ; W_hh]^T   (single K=2048 GEMM)
// 128x128 C tiles, K chunks of 32, LDS double-buffered; tiles are copied with
// async global->LDS (ASYNCcnt) so the copy latency hides under WMMA issue.

constexpr int KC = 32;       // K chunk (one WMMA k-step)
constexpr int LSTR = 40;     // padded LDS stride: 80B rows -> bank spread

__global__ __launch_bounds__(256) void lstm_gate_gemm(
    const bf16* __restrict__ x,    // [512, 1024] embedded slice for step a
    const bf16* __restrict__ h,    // [512, 1024] recurrent state (bf16)
    const bf16* __restrict__ Wih,  // [4096, 1024]
    const bf16* __restrict__ Whh,  // [4096, 1024]
    float* __restrict__ gates) {   // [512, 4096]
  __shared__ bf16 sA[2][128 * LSTR];
  __shared__ bf16 sB[2][128 * LSTR];

  const int tid = threadIdx.x;
  const int lane = tid & 31;
  const int wave = tid >> 5;
  const int mt = blockIdx.x >> 5;   // 4 M tiles
  const int nt = blockIdx.x & 31;   // 32 N tiles
  const int m0 = mt * 128, n0 = nt * 128;
  const int wm = (wave & 3) * 32;
  const int wn = (wave >> 2) * 64;

  const v8f zero8 = {0.f, 0.f, 0.f, 0.f, 0.f, 0.f, 0.f, 0.f};
  v8f acc[2][4];
#pragma unroll
  for (int i = 0; i < 2; ++i)
#pragma unroll
    for (int j = 0; j < 4; ++j) acc[i][j] = zero8;

  // Issue async copies of one 128x32 A tile and one 128x32 B tile into LDS
  // buffer `buf`. 512 16B-chunks per tile / 256 threads = 2 chunks each.
  auto issue = [&](int buf, int k0) {
    const bf16* xs = (k0 < Hn) ? (x + k0) : (h + (k0 - Hn));
    const bf16* wsrc = (k0 < Hn) ? (Wih + k0) : (Whh + (k0 - Hn));
#pragma unroll
    for (int j = 0; j < 2; ++j) {
      const int chunk = tid * 2 + j;  // 0..511
      const int row = chunk >> 2;
      const int c16 = chunk & 3;      // 16B unit within the 64B row
      async_cp16((uint32_t)(size_t)&sA[buf][row * LSTR + c16 * 8],
                 xs + (size_t)(m0 + row) * Hn + c16 * 8);
      async_cp16((uint32_t)(size_t)&sB[buf][row * LSTR + c16 * 8],
                 wsrc + (size_t)(n0 + row) * Hn + c16 * 8);
    }
    // Warm GL2 with weight rows two chunks ahead (global_prefetch_b8).
    const int kp = k0 + 2 * KC;
    if (kp < 2 * Hn) {
      const bf16* wp = (kp < Hn) ? (Wih + kp) : (Whh + (kp - Hn));
      __builtin_prefetch(wp + (size_t)(n0 + (tid >> 1)) * Hn, 0, 3);
    }
  };

  auto compute = [&](int buf) {
    v16bf af[2], bv[4];
#pragma unroll
    for (int i = 0; i < 2; ++i)
      af[i] = frag_ld(&sA[buf][(wm + i * 16) * LSTR], LSTR, lane);
#pragma unroll
    for (int j = 0; j < 4; ++j)
      bv[j] = frag_ld(&sB[buf][(wn + j * 16) * LSTR], LSTR, lane);
#pragma unroll
    for (int i = 0; i < 2; ++i)
#pragma unroll
      for (int j = 0; j < 4; ++j) acc[i][j] = wmma_bf16(af[i], bv[j], acc[i][j]);
  };

  constexpr int NK = (2 * Hn) / KC;  // 64 chunks
  issue(0, 0);
  wait_async0();
  __syncthreads();
#pragma unroll 1
  for (int kc = 0; kc < NK; ++kc) {
    const int cur = kc & 1;
    if (kc + 1 < NK) issue(cur ^ 1, (kc + 1) * KC);  // async fill next buffer
    compute(cur);                                    // WMMA on current buffer
    if (kc + 1 < NK) wait_async0();                  // copies done after math
    __syncthreads();
  }

  const int mhalf = (lane >> 4) ? 8 : 0;
#pragma unroll
  for (int i = 0; i < 2; ++i) {
    const int rbase = m0 + wm + i * 16 + mhalf;
#pragma unroll
    for (int j = 0; j < 4; ++j) {
      const int col = n0 + wn + j * 16 + (lane & 15);
#pragma unroll
      for (int r = 0; r < 8; ++r)
        gates[(size_t)(rbase + r) * G4 + col] = acc[i][j][r];
    }
  }
}

// -------------------- per-step pointwise LSTM update --------------------

__global__ void lstm_pointwise(const float* __restrict__ gates,
                               const float* __restrict__ bias,
                               float* __restrict__ c, bf16* __restrict__ hbf,
                               float* __restrict__ outh,
                               float* __restrict__ outhf,
                               float* __restrict__ outcf, int last) {
  const int idx = blockIdx.x * blockDim.x + threadIdx.x;  // < 512*1024
  const int b = idx >> 10;
  const int hh = idx & 1023;
  const float* g = gates + (size_t)b * G4;
  const float gi = g[hh] + bias[hh];
  const float gf = g[hh + 1024] + bias[hh + 1024];
  const float gg = g[hh + 2048] + bias[hh + 2048];
  const float go = g[hh + 3072] + bias[hh + 3072];
  const float si = 1.0f / (1.0f + __expf(-gi));
  const float sf = 1.0f / (1.0f + __expf(-gf));
  const float so = 1.0f / (1.0f + __expf(-go));
  const float cn = sf * c[idx] + si * tanhf(gg);
  const float hn = so * tanhf(cn);
  c[idx] = cn;
  hbf[idx] = (bf16)hn;  // bf16 state feeds next step's recurrent GEMM
  outh[idx] = hn;       // f32 output slice
  if (last) {
    outhf[idx] = hn;
    outcf[idx] = cn;
  }
}

// -------------------- launch --------------------

extern "C" void kernel_launch(void* const* d_in, const int* in_sizes, int n_in,
                              void* d_out, int out_size, void* d_ws,
                              size_t ws_size, hipStream_t stream) {
  (void)in_sizes; (void)n_in; (void)out_size; (void)ws_size;

  const float* xin  = (const float*)d_in[0];  // inputs [512, 700, 11]
  const float* h0   = (const float*)d_in[1];  // [1, 512, 1024]
  const float* c0   = (const float*)d_in[2];  // [1, 512, 1024]
  const float* Wemb = (const float*)d_in[3];  // [1024, 77]
  const float* bemb = (const float*)d_in[4];  // [1024]
  const float* Wih  = (const float*)d_in[5];  // [4096, 1024]
  const float* Whh  = (const float*)d_in[6];  // [4096, 1024]
  const float* bih  = (const float*)d_in[7];  // [4096]
  const float* bhh  = (const float*)d_in[8];  // [4096]

  // Workspace layout (bytes)
  char* ws = (char*)d_ws;
  bf16*  Wihb  = (bf16*)(ws + 0);          //  8,388,608
  bf16*  Whhb  = (bf16*)(ws + 8388608);    //  8,388,608
  float* bias  = (float*)(ws + 16777216);  //     16,384
  bf16*  hbf   = (bf16*)(ws + 16793600);   //  1,048,576
  float* cst   = (float*)(ws + 17842176);  //  2,097,152
  float* gates = (float*)(ws + 19939328);  //  8,388,608
  bf16*  emb   = (bf16*)(ws + 28327936);   // 104,857,600  (total ~127 MiB)

  float* outh  = (float*)d_out;                    // [100, 512, 1024]
  float* outhf = outh + (size_t)An * Bn * Hn;      // [1, 512, 1024]
  float* outcf = outhf + (size_t)Bn * Hn;          // [1, 512, 1024]

  prep_kernel<<<((size_t)G4 * Hn + 255) / 256, 256, 0, stream>>>(
      Wih, Whh, bih, bhh, h0, c0, Wihb, Whhb, bias, hbf, cst);

  embed_gemm<<<(An * Bn / 128) * (Hn / 128), 256, 0, stream>>>(xin, Wemb, bemb,
                                                               emb);

  for (int a = 0; a < An; ++a) {
    lstm_gate_gemm<<<(Bn / 128) * (G4 / 128), 256, 0, stream>>>(
        emb + (size_t)a * Bn * Hn, hbf, Wihb, Whhb, gates);
    lstm_pointwise<<<(Bn * Hn) / 256, 256, 0, stream>>>(
        gates, bias, cst, hbf, outh + (size_t)a * Bn * Hn, outhf, outcf,
        a == An - 1);
  }
}
